// ConditionExplicitattnBlock_59176059404452
// MI455X (gfx1250) — matrix operand order repaired and checked by reading, not compiled, not run on hardware
//
#include <hip/hip_runtime.h>
#include <hip/hip_bf16.h>
#include <math.h>

typedef __bf16 bf16_t;
typedef __attribute__((ext_vector_type(16))) __bf16 v16bf;
typedef __attribute__((ext_vector_type(8)))  float  v8f;

__device__ inline v8f wmma_bf16f32(v16bf a, v16bf b, v8f c) {
  // D = A(16x32 bf16) * B(32x16 bf16) + C(16x16 f32)
  return __builtin_amdgcn_wmma_f32_16x16x32_bf16(false, a, false, b, (short)0, c,
                                                 false, false);
}

__device__ inline int lane_id() { return (int)(threadIdx.x & 31u); }

// A fragment 16x32 bf16 from row-major [16 x >=32] with leading dim ld (elems).
// Layout: lanes 0-15 hold M=lane, K=0..15;
// lanes 16-31 hold M=lane-16, K=16..31.  One 32B contiguous chunk.
__device__ inline v16bf load_a_frag(const bf16_t* __restrict__ base, int ld) {
  int lane = lane_id();
  int m = lane & 15;
  int kb = (lane >> 4) << 4;  // 0 or 16
  const bf16_t* r0 = base + (size_t)m * ld + kb;
  v16bf r;
#pragma unroll
  for (int i = 0; i < 16; ++i) r[i] = r0[i];
  return r;
}

// B fragment 32x16 from a K-MAJOR tile: src[n][k], ld = K pitch (elems).
// B[k][n] with N = lane&15, K = 16 contiguous values -> one 32B read per lane.
__device__ inline v16bf load_b_kmaj(const bf16_t* __restrict__ base, int ld) {
  int lane = lane_id();
  int n = lane & 15;
  int kb = (lane >> 4) << 4;  // 0 or 16
  const bf16_t* r0 = base + (size_t)n * ld + kb;
  v16bf r;
#pragma unroll
  for (int i = 0; i < 16; ++i) r[i] = r0[i];
  return r;
}

// ---------------------------------------------------------------------------
// f32 -> bf16 conversion
// ---------------------------------------------------------------------------
__global__ void cvt_f32_bf16(const float* __restrict__ in,
                             bf16_t* __restrict__ out, int n) {
  int i = blockIdx.x * blockDim.x + threadIdx.x;
  if (i < n) out[i] = (bf16_t)in[i];
}

// ---------------------------------------------------------------------------
// LayerNorm (f32 in -> bf16 out), one row (D elems) per block.
// ---------------------------------------------------------------------------
__global__ __launch_bounds__(256) void ln_bf16_kernel(
    const float* __restrict__ x, const float* __restrict__ w,
    const float* __restrict__ bvec, bf16_t* __restrict__ out, int Dd) {
  __shared__ float red[256];
  int row = blockIdx.x;
  int tid = threadIdx.x;
  const float* xr = x + (size_t)row * Dd;
  float s = 0.0f;
  for (int i = tid; i < Dd; i += 256) s += xr[i];
  red[tid] = s;
  __syncthreads();
  for (int st = 128; st > 0; st >>= 1) {
    if (tid < st) red[tid] += red[tid + st];
    __syncthreads();
  }
  float mu = red[0] / (float)Dd;
  __syncthreads();
  float s2 = 0.0f;
  for (int i = tid; i < Dd; i += 256) {
    float d = xr[i] - mu;
    s2 += d * d;
  }
  red[tid] = s2;
  __syncthreads();
  for (int st = 128; st > 0; st >>= 1) {
    if (tid < st) red[tid] += red[tid + st];
    __syncthreads();
  }
  float inv = rsqrtf(red[0] / (float)Dd + 1e-6f);
  bf16_t* orow = out + (size_t)row * Dd;
  for (int i = tid; i < Dd; i += 256)
    orow[i] = (bf16_t)((xr[i] - mu) * inv * w[i] + bvec[i]);
}

// ---------------------------------------------------------------------------
// WMMA GEMM: out = act(A[M,K] @ W[N,K]^T + bias) (+ residual)
// Block tile 64(M) x 128(N), K-step 32; 8 waves as 2(M) x 4(N);
// each wave: 32x32 strip = 4 WMMAs per K-step from 2 A-frags x 2 B-frags.
// B is staged K-major (Bs[n][k]) so frag reads are 32B contiguous per lane.
// ---------------------------------------------------------------------------
#define GT_M 64
#define GT_N 128
#define GT_K 32
#define KP (GT_K + 8)   // LDS K pitch: 40 elems = 80B (16B aligned, 20 banks)

__global__ __launch_bounds__(256) void gemm_bf16_kernel(
    const bf16_t* __restrict__ A, const bf16_t* __restrict__ W,
    const float* __restrict__ bias, const float* __restrict__ residual,
    float* __restrict__ outF, bf16_t* __restrict__ outB,
    int M, int N, int K, int act) {
  __shared__ bf16_t As[GT_M][KP];
  __shared__ bf16_t Bs[GT_N][KP];
  int tid = threadIdx.x;
  int lane = tid & 31;
  int wave = tid >> 5;
  int wm = wave & 1;   // 0..1 (M sub-tile of 32)
  int wn = wave >> 1;  // 0..3 (N sub-tile of 32)
  int m0 = blockIdx.y * GT_M;
  int n0 = blockIdx.x * GT_N;

  // A staging: 64 rows x 32 k, 8 contiguous bf16 per thread
  int am = tid & 63;
  int ak = (tid >> 6) << 3;  // 0,8,16,24
  // B staging: 128 rows x 32 k, 16 contiguous bf16 per thread (row copy of W)
  int bn = tid >> 1;        // 0..127
  int bk = (tid & 1) << 4;  // 0 or 16

  v8f acc00 = {}, acc01 = {}, acc10 = {}, acc11 = {};

  const bf16_t* Arow = A + (size_t)(m0 + am) * K + ak;
  const bf16_t* Wrow = W + (size_t)(n0 + bn) * K + bk;

  for (int k0 = 0; k0 < K; k0 += GT_K) {
#pragma unroll
    for (int i = 0; i < 8; ++i) As[am][ak + i] = Arow[k0 + i];
#pragma unroll
    for (int i = 0; i < 16; ++i) Bs[bn][bk + i] = Wrow[k0 + i];
    if (k0 + GT_K < K) {
      __builtin_prefetch(Arow + k0 + GT_K, 0, 1);  // global_prefetch_b8
      __builtin_prefetch(Wrow + k0 + GT_K, 0, 1);
    }
    __syncthreads();
    v16bf a0 = load_a_frag(&As[wm * 32][0], KP);
    v16bf a1 = load_a_frag(&As[wm * 32 + 16][0], KP);
    v16bf b0 = load_b_kmaj(&Bs[wn * 32][0], KP);
    v16bf b1 = load_b_kmaj(&Bs[wn * 32 + 16][0], KP);
    acc00 = wmma_bf16f32(a0, b0, acc00);
    acc01 = wmma_bf16f32(a0, b1, acc01);
    acc10 = wmma_bf16f32(a1, b0, acc10);
    acc11 = wmma_bf16f32(a1, b1, acc11);
    __syncthreads();
  }

  int lh = lane >> 4;
  int nl = lane & 15;
  v8f accs[2][2];
  accs[0][0] = acc00; accs[0][1] = acc01;
  accs[1][0] = acc10; accs[1][1] = acc11;
#pragma unroll
  for (int mi = 0; mi < 2; ++mi) {
#pragma unroll
    for (int nj = 0; nj < 2; ++nj) {
#pragma unroll
      for (int v = 0; v < 8; ++v) {
        int m = m0 + wm * 32 + mi * 16 + v + (lh << 3);
        int n = n0 + wn * 32 + nj * 16 + nl;
        float val = accs[mi][nj][v];
        if (bias) val += bias[n];
        if (act == 1)  // exact GELU
          val = 0.5f * val * (1.0f + erff(val * 0.7071067811865475f));
        if (residual) val += residual[(size_t)m * N + n];
        if (outF) outF[(size_t)m * N + n] = val;
        if (outB) outB[(size_t)m * N + n] = (bf16_t)val;
      }
    }
  }
}

// ---------------------------------------------------------------------------
// Flash attention with additive distance bias, all-WMMA.
// Grid: (Sq/128, H, B); 8 waves/block; one wave = one 16-row Q tile of head h.
// K/V tiles for the block's (b,h) are staged cooperatively in LDS once per
// 32-wide KV step: K row-copied (K-contiguous frags for QK^T), V transposed
// (Vs[d][kv] -> K-contiguous frags for P@V). Out may alias Q.
// ---------------------------------------------------------------------------
__global__ __launch_bounds__(256) void attn_bias_kernel(
    const bf16_t* __restrict__ Q, const bf16_t* __restrict__ Kt,
    const bf16_t* __restrict__ Vt, const float* __restrict__ dist,
    const float* __restrict__ gamma_p, bf16_t* __restrict__ Out,
    int Sq, int Skv, int Dd) {
  __shared__ bf16_t Ks[32][72];       // [kv][d], pitch 144B (36 banks)
  __shared__ bf16_t Vs[64][40];       // [d][kv] transposed, pitch 80B
  __shared__ bf16_t Plds[8][16][40];  // per-wave P bounce (16x32 + pad)
  int tid = threadIdx.x;
  int lane = tid & 31;
  int wave = tid >> 5;
  int b = blockIdx.z;
  int h = blockIdx.y;
  int q0 = blockIdx.x * 128 + wave * 16;
  int lh = lane >> 4;
  int nl = lane & 15;
  float g = gamma_p[0];
  float bscale = -(g * g);
  const float scale = 0.125f;  // 1/sqrt(64)

  const bf16_t* qbase = Q + ((size_t)(b * Sq + q0)) * Dd + h * 64;
  v16bf qf0 = load_a_frag(qbase, Dd);       // d = 0..31
  v16bf qf1 = load_a_frag(qbase + 32, Dd);  // d = 32..63

  v8f o0 = {}, o1 = {}, o2 = {}, o3 = {};
  float mrow[8], lrow[8];
#pragma unroll
  for (int v = 0; v < 8; ++v) {
    mrow[v] = -1e30f;
    lrow[v] = 0.0f;
  }

  // staging indices: 32 kv rows x 64 d, 8 contiguous d per thread
  int tr = tid >> 3;        // 0..31 (kv row)
  int tc = (tid & 7) << 3;  // 0..56 (d col)

  const float* drow = dist + ((size_t)(b * Sq + q0 + (lh << 3))) * Skv + nl;

  for (int kv0 = 0; kv0 < Skv; kv0 += 32) {
    // ---- cooperative stage of K (row copy) and V (transposed) ----
    const bf16_t* kr = Kt + ((size_t)(b * Skv + kv0 + tr)) * Dd + h * 64 + tc;
    const bf16_t* vr = Vt + ((size_t)(b * Skv + kv0 + tr)) * Dd + h * 64 + tc;
#pragma unroll
    for (int i = 0; i < 8; ++i) Ks[tr][tc + i] = kr[i];
#pragma unroll
    for (int i = 0; i < 8; ++i) Vs[tc + i][tr] = vr[i];
    __syncthreads();

    // ---- scores for two 16-col subtiles (QK^T via WMMA) ----
    v8f s0 = {}, s1 = {};
    s0 = wmma_bf16f32(qf0, load_b_kmaj(&Ks[0][0], 72), s0);
    s0 = wmma_bf16f32(qf1, load_b_kmaj(&Ks[0][32], 72), s0);
    s1 = wmma_bf16f32(qf0, load_b_kmaj(&Ks[16][0], 72), s1);
    s1 = wmma_bf16f32(qf1, load_b_kmaj(&Ks[16][32], 72), s1);

    // ---- scale + additive distance bias ----
#pragma unroll
    for (int v = 0; v < 8; ++v) {
      float d0 = drow[(size_t)v * Skv + kv0];
      float d1 = drow[(size_t)v * Skv + kv0 + 16];
      s0[v] = s0[v] * scale + d0 * bscale;
      s1[v] = s1[v] * scale + d1 * bscale;
    }

    // ---- online softmax (row reductions across 16-lane halves) ----
#pragma unroll
    for (int v = 0; v < 8; ++v) {
      float mx = fmaxf(s0[v], s1[v]);
      mx = fmaxf(mx, __shfl_xor(mx, 1, 32));
      mx = fmaxf(mx, __shfl_xor(mx, 2, 32));
      mx = fmaxf(mx, __shfl_xor(mx, 4, 32));
      mx = fmaxf(mx, __shfl_xor(mx, 8, 32));
      float mn = fmaxf(mrow[v], mx);
      float corr = __expf(mrow[v] - mn);
      float p0 = __expf(s0[v] - mn);
      float p1 = __expf(s1[v] - mn);
      float rs = p0 + p1;
      rs += __shfl_xor(rs, 1, 32);
      rs += __shfl_xor(rs, 2, 32);
      rs += __shfl_xor(rs, 4, 32);
      rs += __shfl_xor(rs, 8, 32);
      lrow[v] = lrow[v] * corr + rs;
      mrow[v] = mn;
      s0[v] = p0;
      s1[v] = p1;
      o0[v] *= corr;
      o1[v] *= corr;
      o2[v] *= corr;
      o3[v] *= corr;
    }

    // ---- C-layout P -> A-layout via per-wave LDS bounce ----
#pragma unroll
    for (int v = 0; v < 8; ++v) {
      int m = v + (lh << 3);
      Plds[wave][m][nl] = (bf16_t)s0[v];
      Plds[wave][m][nl + 16] = (bf16_t)s1[v];
    }
    asm volatile("s_wait_dscnt 0" ::: "memory");  // wave-local LDS RAW
    v16bf pf = load_a_frag(&Plds[wave][0][0], 40);

    // ---- O += P @ V (four 16-wide N chunks of HD=64) ----
    o0 = wmma_bf16f32(pf, load_b_kmaj(&Vs[0][0], 40), o0);
    o1 = wmma_bf16f32(pf, load_b_kmaj(&Vs[16][0], 40), o1);
    o2 = wmma_bf16f32(pf, load_b_kmaj(&Vs[32][0], 40), o2);
    o3 = wmma_bf16f32(pf, load_b_kmaj(&Vs[48][0], 40), o3);
    __syncthreads();  // protect Ks/Vs before next stage
  }

  bf16_t* orow = Out + ((size_t)(b * Sq + q0 + (lh << 3))) * Dd + h * 64 + nl;
#pragma unroll
  for (int v = 0; v < 8; ++v) {
    float inv = 1.0f / lrow[v];
    orow[(size_t)v * Dd + 0] = (bf16_t)(o0[v] * inv);
    orow[(size_t)v * Dd + 16] = (bf16_t)(o1[v] * inv);
    orow[(size_t)v * Dd + 32] = (bf16_t)(o2[v] * inv);
    orow[(size_t)v * Dd + 48] = (bf16_t)(o3[v] * inv);
  }
}

// ---------------------------------------------------------------------------
// Host orchestration
// ---------------------------------------------------------------------------
extern "C" void kernel_launch(void* const* d_in, const int* in_sizes, int n_in,
                              void* d_out, int out_size, void* d_ws,
                              size_t ws_size, hipStream_t stream) {
  (void)in_sizes; (void)n_in; (void)out_size; (void)ws_size;
  constexpr int Bb = 2, S = 2048, L = 1024, D = 1024, C = 768, Hn = 16,
                DF = 4096;
  constexpr int MX = Bb * S;  // 4096
  constexpr int MC = Bb * L;  // 2048

  const float* x        = (const float*)d_in[0];
  const float* cond     = (const float*)d_in[1];
  const float* sa_dist  = (const float*)d_in[2];
  const float* ca_dist  = (const float*)d_in[3];
  const float* gamma_ca = (const float*)d_in[4];
  const float* gamma_sa = (const float*)d_in[5];
  const float* ln1_w = (const float*)d_in[6];
  const float* ln1_b = (const float*)d_in[7];
  const float* ln2_w = (const float*)d_in[8];
  const float* ln2_b = (const float*)d_in[9];
  const float* ln3_w = (const float*)d_in[10];
  const float* ln3_b = (const float*)d_in[11];
  const float* ca_wq = (const float*)d_in[12];
  const float* ca_wk = (const float*)d_in[13];
  const float* ca_wv = (const float*)d_in[14];
  const float* ca_wo = (const float*)d_in[15];
  const float* sa_wq = (const float*)d_in[16];
  const float* sa_wk = (const float*)d_in[17];
  const float* sa_wv = (const float*)d_in[18];
  const float* sa_wo = (const float*)d_in[19];
  const float* mlp_w1 = (const float*)d_in[20];
  const float* mlp_b1 = (const float*)d_in[21];
  const float* mlp_w2 = (const float*)d_in[22];
  const float* mlp_b2 = (const float*)d_in[23];

  char* wsp = (char*)d_ws;
  size_t off = 0;
  auto carve = [&](size_t bytes) -> void* {
    void* p = (void*)(wsp + off);
    off += (bytes + 255) & ~(size_t)255;
    return p;
  };

  bf16_t* b_ca_wq = (bf16_t*)carve((size_t)D * D * 2);
  bf16_t* b_ca_wk = (bf16_t*)carve((size_t)D * C * 2);
  bf16_t* b_ca_wv = (bf16_t*)carve((size_t)D * C * 2);
  bf16_t* b_ca_wo = (bf16_t*)carve((size_t)D * D * 2);
  bf16_t* b_sa_wq = (bf16_t*)carve((size_t)D * D * 2);
  bf16_t* b_sa_wk = (bf16_t*)carve((size_t)D * D * 2);
  bf16_t* b_sa_wv = (bf16_t*)carve((size_t)D * D * 2);
  bf16_t* b_sa_wo = (bf16_t*)carve((size_t)D * D * 2);
  bf16_t* b_w1 = (bf16_t*)carve((size_t)DF * D * 2);
  bf16_t* b_w2 = (bf16_t*)carve((size_t)D * DF * 2);
  bf16_t* b_ln = (bf16_t*)carve((size_t)MX * D * 2);
  bf16_t* b_cond = (bf16_t*)carve((size_t)MC * C * 2);
  bf16_t* b_q = (bf16_t*)carve((size_t)MX * D * 2);
  bf16_t* b_k = (bf16_t*)carve((size_t)MX * D * 2);
  bf16_t* b_v = (bf16_t*)carve((size_t)MX * D * 2);
  float* f_x1 = (float*)carve((size_t)MX * D * 4);
  float* f_x2 = (float*)carve((size_t)MX * D * 4);
  bf16_t* b_h = (bf16_t*)carve((size_t)MX * DF * 2);

  auto cvt = [&](const float* src, bf16_t* dst, int n) {
    cvt_f32_bf16<<<(n + 255) / 256, 256, 0, stream>>>(src, dst, n);
  };
  auto gemm = [&](const bf16_t* A, const bf16_t* W, const float* bias,
                  const float* resid, float* oF, bf16_t* oB, int M, int N,
                  int K, int act) {
    dim3 g(N / GT_N, M / GT_M);
    gemm_bf16_kernel<<<g, 256, 0, stream>>>(A, W, bias, resid, oF, oB, M, N, K,
                                            act);
  };

  // Weight / cond conversion to bf16
  cvt(ca_wq, b_ca_wq, D * D);
  cvt(ca_wk, b_ca_wk, D * C);
  cvt(ca_wv, b_ca_wv, D * C);
  cvt(ca_wo, b_ca_wo, D * D);
  cvt(sa_wq, b_sa_wq, D * D);
  cvt(sa_wk, b_sa_wk, D * D);
  cvt(sa_wv, b_sa_wv, D * D);
  cvt(sa_wo, b_sa_wo, D * D);
  cvt(mlp_w1, b_w1, DF * D);
  cvt(mlp_w2, b_w2, D * DF);
  cvt(cond, b_cond, MC * C);

  // ---- Cross-attention ----
  ln_bf16_kernel<<<MX, 256, 0, stream>>>(x, ln1_w, ln1_b, b_ln, D);
  gemm(b_ln, b_ca_wq, nullptr, nullptr, nullptr, b_q, MX, D, D, 0);
  gemm(b_cond, b_ca_wk, nullptr, nullptr, nullptr, b_k, MC, D, C, 0);
  gemm(b_cond, b_ca_wv, nullptr, nullptr, nullptr, b_v, MC, D, C, 0);
  attn_bias_kernel<<<dim3(S / 128, Hn, Bb), 256, 0, stream>>>(
      b_q, b_k, b_v, ca_dist, gamma_ca, b_q, S, L, D);
  gemm(b_q, b_ca_wo, nullptr, x, f_x1, nullptr, MX, D, D, 0);

  // ---- Self-attention ----
  ln_bf16_kernel<<<MX, 256, 0, stream>>>(f_x1, ln2_w, ln2_b, b_ln, D);
  gemm(b_ln, b_sa_wq, nullptr, nullptr, nullptr, b_q, MX, D, D, 0);
  gemm(b_ln, b_sa_wk, nullptr, nullptr, nullptr, b_k, MX, D, D, 0);
  gemm(b_ln, b_sa_wv, nullptr, nullptr, nullptr, b_v, MX, D, D, 0);
  attn_bias_kernel<<<dim3(S / 128, Hn, Bb), 256, 0, stream>>>(
      b_q, b_k, b_v, sa_dist, gamma_sa, b_q, S, S, D);
  gemm(b_q, b_sa_wo, nullptr, f_x1, f_x2, nullptr, MX, D, D, 0);

  // ---- MLP ----
  ln_bf16_kernel<<<MX, 256, 0, stream>>>(f_x2, ln3_w, ln3_b, b_ln, D);
  gemm(b_ln, b_w1, mlp_b1, nullptr, nullptr, b_h, MX, DF, D, 1);
  gemm(b_h, b_w2, mlp_b2, f_x2, (float*)d_out, nullptr, MX, D, DF, 0);
}
